// DiffusionUnit_12549894439613
// MI455X (gfx1250) — matrix operand order repaired and checked by reading, not compiled, not run on hardware
//
#include <hip/hip_runtime.h>
#include <hip/hip_bf16.h>

#define NPTS 65536
#define CH   256
#define NS   16
#define BN_EPS 1e-5f

typedef __attribute__((ext_vector_type(16))) _Float16 v16h;
typedef __attribute__((ext_vector_type(8)))  _Float16 v8h;
typedef __attribute__((ext_vector_type(4)))  _Float16 v4h;
typedef __attribute__((ext_vector_type(8)))  float    v8f;
typedef __attribute__((ext_vector_type(4)))  int      v4i;

#if defined(__HIP_DEVICE_COMPILE__)
#if __has_builtin(__builtin_amdgcn_global_load_async_to_lds_b128)
#define HAVE_ASYNC_LDS 1
#endif
#endif

typedef __attribute__((address_space(1))) v4i glob_v4i;
typedef __attribute__((address_space(3))) v4i lds_v4i;

__device__ __forceinline__ void wait_async0() {
#if defined(HAVE_ASYNC_LDS)
#if __has_builtin(__builtin_amdgcn_s_wait_asynccnt)
    __builtin_amdgcn_s_wait_asynccnt(0);
#else
    asm volatile("s_wait_asynccnt 0x0" ::: "memory");
#endif
#endif
}

// ---------------------------------------------------------------------------
// prep: convert both weight matrices fp32 -> f16, zero BN accumulators
// ---------------------------------------------------------------------------
__global__ void prep_kernel(const float* __restrict__ Wpre, const float* __restrict__ Wv,
                            _Float16* __restrict__ WpreH, _Float16* __restrict__ WvH,
                            float* __restrict__ sums, float* __restrict__ sumsq) {
    int i = blockIdx.x * blockDim.x + threadIdx.x;
    if (i < CH * CH) {
        WpreH[i] = (_Float16)Wpre[i];
        WvH[i]   = (_Float16)Wv[i];
    }
    if (i < CH) { sums[i] = 0.0f; sumsq[i] = 0.0f; }
}

// ---------------------------------------------------------------------------
// WMMA GEMM: Out[m,n] = sum_k A[m,k] * W[n,k] + bias[n]
//   W : [256,256] f16 row-major (row = output channel)
// AFP32 1: A is fp32 (u), converted to f16 while staging into LDS
// AFP32 0: A is f16 (m)
// MODE 0: store f16 (u_pre)
// MODE 1: store f32 (h) + accumulate per-column sum/sumsq for BN
// Block: 128 threads (4 waves); tile 128(M) x 64(N); K streamed in steps of 32.
// Per wave: 2 M-fragments x 4 N-subtiles = 8 WMMAs per K step (B frags reused).
// ---------------------------------------------------------------------------
template <int MODE, int AFP32>
__global__ __launch_bounds__(128)
void gemm_kernel(const void* __restrict__ Ain, const _Float16* __restrict__ W,
                 const float* __restrict__ bias,
                 _Float16* __restrict__ outH, float* __restrict__ outF,
                 float* __restrict__ sums, float* __restrict__ sumsq) {
    constexpr int TM = 128, TN = 64;
    constexpr int BPAD = 8, APAD = 8;
    // Bs[n][k]: row stride 264 halves = 132 dwords -> 4-bank rotation per row;
    // 16B fragment loads hit disjoint banks across lanes 0..15.
    __shared__ __align__(16) _Float16 Bs[TN][CH + BPAD];
    // As[m][k]: row stride 40 halves = 20 dwords -> conflict-free 16B loads.
    __shared__ __align__(16) _Float16 As[TM][32 + APAD];

    const int m0   = blockIdx.x * TM;
    const int n0   = blockIdx.y * TN;
    const int tid  = threadIdx.x;
    const int wave = tid >> 5;
    const int lane = tid & 31;

    // ---- stage the 64-column weight panel (full K, 32KB) into LDS as [n][k]
#if defined(HAVE_ASYNC_LDS)
    for (int it = tid; it < TN * (CH / 8); it += 128) {
        int n  = it & (TN - 1);
        int kc = it >> 6;                       // 0..31 chunk of 8 halves
        __builtin_amdgcn_global_load_async_to_lds_b128(
            (glob_v4i*)&W[(size_t)(n0 + n) * CH + kc * 8],
            (lds_v4i*)&Bs[n][kc * 8], 0, 0);
    }
    wait_async0();
#else
    for (int it = tid; it < TN * (CH / 8); it += 128) {
        int n  = it & (TN - 1);
        int kc = it >> 6;
        float4 v = *reinterpret_cast<const float4*>(&W[(size_t)(n0 + n) * CH + kc * 8]);
        *reinterpret_cast<float4*>(&Bs[n][kc * 8]) = v;
    }
#endif

    v8f acc[2][4];
#pragma unroll
    for (int a = 0; a < 2; ++a)
#pragma unroll
        for (int b = 0; b < 4; ++b) acc[a][b] = v8f{};

    const int mrow = lane & 15;
    const int hi   = lane >> 4;        // 0 or 1
    const int ksA  = hi * 8;           // A frag K subset {0..7,16..23}/{8..15,24..31}
    const int ksB  = hi * 8;           // B frag K subset {0..7,16..23}/{8..15,24..31}
    const int ncol = lane & 15;

    for (int k0 = 0; k0 < CH; k0 += 32) {
        __syncthreads();
        // ---- stage A tile 128x32: thread t owns row t (32 halves)
        if (AFP32) {
            const float* src = (const float*)Ain + (size_t)(m0 + tid) * CH + k0;
#pragma unroll
            for (int c = 0; c < 4; ++c) {
                float4 x = *reinterpret_cast<const float4*>(src + c * 8);
                float4 y = *reinterpret_cast<const float4*>(src + c * 8 + 4);
                v8h p;
                p[0] = (_Float16)x.x; p[1] = (_Float16)x.y;
                p[2] = (_Float16)x.z; p[3] = (_Float16)x.w;
                p[4] = (_Float16)y.x; p[5] = (_Float16)y.y;
                p[6] = (_Float16)y.z; p[7] = (_Float16)y.w;
                *reinterpret_cast<v8h*>(&As[tid][c * 8]) = p;
            }
        } else {
            const _Float16* src = (const _Float16*)Ain + (size_t)(m0 + tid) * CH + k0;
#pragma unroll
            for (int c = 0; c < 4; ++c) {
                float4 v = *reinterpret_cast<const float4*>(src + c * 8);
                *reinterpret_cast<float4*>(&As[tid][c * 8]) = v;
            }
        }
        __syncthreads();

        // ---- A fragments (documented 16-bit A layout), 2 per wave
        v16h afrag[2];
#pragma unroll
        for (int ms = 0; ms < 2; ++ms) {
            const _Float16* arow = &As[wave * 32 + ms * 16 + mrow][0];
            union { v16h v; v8h h[2]; } au;
            au.h[0] = *reinterpret_cast<const v8h*>(&arow[ksA]);
            au.h[1] = *reinterpret_cast<const v8h*>(&arow[16 + ksA]);
            afrag[ms] = au.v;
        }

#pragma unroll
        for (int ns = 0; ns < 4; ++ns) {
            const _Float16* brow = &Bs[ns * 16 + ncol][k0 + ksB];
            union { v16h v; v8h h[2]; } bu;
            bu.h[0] = *reinterpret_cast<const v8h*>(brow);
            bu.h[1] = *reinterpret_cast<const v8h*>(brow + 16);
#pragma unroll
            for (int ms = 0; ms < 2; ++ms) {
                acc[ms][ns] = __builtin_amdgcn_wmma_f32_16x16x32_f16(
                    false, afrag[ms], false, bu.v, (short)0, acc[ms][ns], false, false);
            }
        }
    }

    // ---- epilogue: C/D layout — lane gives N, vgpr r gives M = hi*8 + r
#pragma unroll
    for (int ns = 0; ns < 4; ++ns) {
        int n = n0 + ns * 16 + ncol;
        float b = bias[n];
        float s = 0.0f, s2 = 0.0f;
#pragma unroll
        for (int ms = 0; ms < 2; ++ms) {
#pragma unroll
            for (int r = 0; r < 8; ++r) {
                int m = m0 + wave * 32 + ms * 16 + hi * 8 + r;
                float v = acc[ms][ns][r] + b;
                if (MODE == 0) {
                    outH[(size_t)m * CH + n] = (_Float16)v;
                } else {
                    outF[(size_t)m * CH + n] = v;
                    s += v; s2 += v * v;
                }
            }
        }
        if (MODE == 1) {
            // fold half-wave partners (same column, different rows)
            s  += __shfl_xor(s, 16, 32);
            s2 += __shfl_xor(s2, 16, 32);
            if (hi == 0) {
                atomicAdd(&sums[n], s);
                atomicAdd(&sumsq[n], s2);
            }
        }
    }
}

// ---------------------------------------------------------------------------
// diffuse: m[i,c] = (1/16) * sum_j relu(u_pre[idx[i,j],c] - u_pre[i,c])
// u_pre is L2-resident f16; 64 threads per row, 4 channels per thread.
// ---------------------------------------------------------------------------
__global__ __launch_bounds__(256)
void diffuse_kernel(const _Float16* __restrict__ upre, const int* __restrict__ idx,
                    _Float16* __restrict__ mh) {
    int row = blockIdx.x * 4 + (threadIdx.x >> 6);
    int cg  = threadIdx.x & 63;                      // channel group of 4
    const size_t coff = (size_t)cg * 4;
    v4h cv = *reinterpret_cast<const v4h*>(&upre[(size_t)row * CH + coff]);
    float c0 = (float)cv[0], c1 = (float)cv[1], c2 = (float)cv[2], c3 = (float)cv[3];
    float a0 = 0.f, a1 = 0.f, a2 = 0.f, a3 = 0.f;
    const int* ip = &idx[row * NS];
#pragma unroll
    for (int j = 0; j < NS; ++j) {
        int nb = ip[j];
        v4h nv = *reinterpret_cast<const v4h*>(&upre[(size_t)nb * CH + coff]);
        a0 += fmaxf(0.f, (float)nv[0] - c0);
        a1 += fmaxf(0.f, (float)nv[1] - c1);
        a2 += fmaxf(0.f, (float)nv[2] - c2);
        a3 += fmaxf(0.f, (float)nv[3] - c3);
    }
    v4h o;
    o[0] = (_Float16)(a0 * 0.0625f);
    o[1] = (_Float16)(a1 * 0.0625f);
    o[2] = (_Float16)(a2 * 0.0625f);
    o[3] = (_Float16)(a3 * 0.0625f);
    *reinterpret_cast<v4h*>(&mh[(size_t)row * CH + coff]) = o;
}

// ---------------------------------------------------------------------------
// bn_stats: fold sums into per-channel scale/shift
// ---------------------------------------------------------------------------
__global__ void bn_stats_kernel(const float* __restrict__ sums, const float* __restrict__ sumsq,
                                const float* __restrict__ gamma, const float* __restrict__ beta,
                                float* __restrict__ scale, float* __restrict__ shift) {
    int c = threadIdx.x;
    float invN = 1.0f / (float)NPTS;
    float mu   = sums[c] * invN;
    float var  = sumsq[c] * invN - mu * mu;
    float inv  = rsqrtf(var + BN_EPS);
    float sc   = gamma[c] * inv;
    scale[c] = sc;
    shift[c] = beta[c] - mu * sc;
}

// ---------------------------------------------------------------------------
// finalize: out = relu(h * scale + shift) + u   (vectorized 4-wide)
// ---------------------------------------------------------------------------
__global__ void finalize_kernel(const float* __restrict__ h, const float* __restrict__ u,
                                const float* __restrict__ scale, const float* __restrict__ shift,
                                float* __restrict__ out) {
    size_t i = (size_t)blockIdx.x * blockDim.x + threadIdx.x;   // over N*C/4
    int c = (int)(i & 63) * 4;                                  // first of 4 channels
    float4 hv = reinterpret_cast<const float4*>(h)[i];
    float4 uv = reinterpret_cast<const float4*>(u)[i];
    float4 sc = *reinterpret_cast<const float4*>(&scale[c]);
    float4 sh = *reinterpret_cast<const float4*>(&shift[c]);
    float4 o;
    o.x = fmaxf(0.f, hv.x * sc.x + sh.x) + uv.x;
    o.y = fmaxf(0.f, hv.y * sc.y + sh.y) + uv.y;
    o.z = fmaxf(0.f, hv.z * sc.z + sh.z) + uv.z;
    o.w = fmaxf(0.f, hv.w * sc.w + sh.w) + uv.w;
    reinterpret_cast<float4*>(out)[i] = o;
}

// ---------------------------------------------------------------------------
extern "C" void kernel_launch(void* const* d_in, const int* in_sizes, int n_in,
                              void* d_out, int out_size, void* d_ws, size_t ws_size,
                              hipStream_t stream) {
    (void)in_sizes; (void)n_in; (void)out_size; (void)ws_size;
    // inputs: p(0) u(1) o(2) idx(3) W_pre(4) b_pre(5) W_v(6) b_v(7) gamma(8) beta(9)
    const float* u     = (const float*)d_in[1];
    const int*   idx   = (const int*)d_in[3];
    const float* W_pre = (const float*)d_in[4];
    const float* b_pre = (const float*)d_in[5];
    const float* W_v   = (const float*)d_in[6];
    const float* b_v   = (const float*)d_in[7];
    const float* gamma = (const float*)d_in[8];
    const float* beta  = (const float*)d_in[9];
    float* out = (float*)d_out;

    char* ws = (char*)d_ws;
    const size_t NC = (size_t)NPTS * CH;
    // u_pre_h occupies [0,32MB) and is dead after diffuse; h (f32, 64MB)
    // aliases [0,64MB). m_h lives at [64MB,96MB) so gemm2 (reads m_h,
    // writes h) has no overlap.
    _Float16* u_pre_h = (_Float16*)(ws);                    // 32MB
    float*    h       = (float*)(ws);                       // 64MB (later)
    _Float16* m_h     = (_Float16*)(ws + NC * 4);           // 32MB
    char* tail = ws + NC * 4 + NC * 2;
    _Float16* Wh_pre  = (_Float16*)(tail);                  // 128KB
    _Float16* Wh_v    = (_Float16*)(tail + (size_t)CH * CH * 2);
    float*    sums    = (float*)(tail + (size_t)CH * CH * 4);
    float*    sumsq   = sums + CH;
    float*    scale   = sums + 2 * CH;
    float*    shift   = sums + 3 * CH;

    // 1) weights -> f16, zero BN accumulators
    prep_kernel<<<(CH * CH + 255) / 256, 256, 0, stream>>>(W_pre, W_v, Wh_pre, Wh_v, sums, sumsq);
    // 2) u_pre = u @ W_pre^T + b_pre (fp32 A converted in-kernel, f16 out)
    dim3 gg(NPTS / 128, CH / 64);
    gemm_kernel<0, 1><<<gg, 128, 0, stream>>>(u, Wh_pre, b_pre, u_pre_h, nullptr, nullptr, nullptr);
    // 3) diffusion gather/mean (f16 out)
    diffuse_kernel<<<NPTS / 4, 256, 0, stream>>>(u_pre_h, idx, m_h);
    // 4) h = m @ W_v^T + b_v (f32 out) + BN column stats
    gemm_kernel<1, 0><<<gg, 128, 0, stream>>>(m_h, Wh_v, b_v, nullptr, h, sums, sumsq);
    // 5) BN stats -> scale/shift
    bn_stats_kernel<<<1, CH, 0, stream>>>(sums, sumsq, gamma, beta, scale, shift);
    // 6) out = relu(BN(h)) + u
    finalize_kernel<<<(int)(NC / 4 / 256), 256, 0, stream>>>(h, u, scale, shift, out);
}
